// MultiHeadAttention_37529424232825
// MI455X (gfx1250) — compile-verified, matrix operand order checked by reading, not run on hardware
//
#include <hip/hip_runtime.h>
#include <hip/hip_bf16.h>
#include <stdint.h>

// Problem constants
#define BB 2
#define SS 2048
#define EE 1024
#define HH 16
#define HD 64

typedef unsigned short u16;
typedef __attribute__((ext_vector_type(4)))  unsigned int u32x4;
typedef __attribute__((ext_vector_type(4)))  unsigned int uint32x4;
typedef __attribute__((ext_vector_type(4)))  int          int32x4;
typedef __attribute__((ext_vector_type(8)))  int          int32x8;
typedef __attribute__((ext_vector_type(16))) __bf16       v16bf;
typedef __attribute__((ext_vector_type(8)))  float        v8f;

#if defined(__has_builtin)
#if __has_builtin(__builtin_amdgcn_tensor_load_to_lds) && \
    __has_builtin(__builtin_amdgcn_s_wait_tensorcnt)
#define HAVE_TDM 1
#endif
#endif
#ifndef HAVE_TDM
#define HAVE_TDM 0
#endif

union FragBF { v16bf v; u32x4 q[2]; };

// float -> bf16 (round to nearest even)
__device__ __forceinline__ u16 f2bf(float x) {
  union { float f; unsigned int u; } t; t.f = x;
  unsigned int r = t.u + 0x7fffu + ((t.u >> 16) & 1u);
  return (u16)(r >> 16);
}

// Load one 16x32 bf16 WMMA operand fragment from a (Rows, K) row-major matrix
// (global or LDS). A operand: rows = M.  B operand: matrix stored as (N, K)
// row-major (true for w_qkv, w_out, k', v^T and the staged LDS tiles).
// ISA 16-bit layout: lanes 0-15 hold K {k0..k0+7, k0+16..k0+23},
//                    lanes 16-31 hold K {k0+8..k0+15, k0+24..k0+31}.
__device__ __forceinline__ v16bf frag_ld(const u16* base, int row, int ld,
                                         int k0, int lane) {
  const int half = (lane >> 4) & 1;
  const u16* p = base + (size_t)(row + (lane & 15)) * (size_t)ld + (k0 + half * 8);
  FragBF f;
  f.q[0] = *(const u32x4*)(p);
  f.q[1] = *(const u32x4*)(p + 16);
  return f.v;
}

__device__ __forceinline__ v8f wmma_bf16(v16bf a, v16bf b, v8f c) {
  return __builtin_amdgcn_wmma_f32_16x16x32_bf16(false, a, false, b,
                                                 (short)0, c, false, false);
}

#if HAVE_TDM
// Issue one TDM 2D tile load: (tile_d1 rows x tile_d0 elems, bf16) from a
// row-major tensor with row stride `stride0` (elements) into LDS at lds_off.
// D# layout per CDNA5 ISA (group0 128b, group1 256b).
__device__ __forceinline__ void tdm_load_2d(const u16* gtile, unsigned lds_off,
                                            unsigned tensor_d0, unsigned tensor_d1,
                                            unsigned tile_d0, unsigned tile_d1,
                                            unsigned stride0) {
  const unsigned long long ga = (unsigned long long)(uintptr_t)gtile;
  uint32x4 g0;
  g0.x = 1u;                                          // count=1, user mode
  g0.y = lds_off;                                     // lds_addr [63:32]
  g0.z = (unsigned)(ga & 0xffffffffu);                // global_addr [95:64]
  g0.w = (unsigned)((ga >> 32) & 0x01ffffffu) | (2u << 30);  // addr[56:32]|type=2
  int32x8 g1;
  g1[0] = (int)(1u << 16);                            // data_size=1 (2 bytes)
  g1[1] = (int)((tensor_d0 & 0xffffu) << 16);         // tensor_dim0[15:0]
  g1[2] = (int)(((tensor_d0 >> 16) & 0xffffu) |
                ((tensor_d1 & 0xffffu) << 16));       // td0[31:16] | td1[15:0]
  g1[3] = (int)(((tensor_d1 >> 16) & 0xffffu) |
                ((tile_d0 & 0xffffu) << 16));         // td1[31:16] | tile_dim0
  g1[4] = (int)(tile_d1 & 0xffffu);                   // tile_dim1 (tile_dim2=0)
  g1[5] = (int)stride0;                               // tensor_dim0_stride lo32
  g1[6] = 0;                                          // stride0 hi / stride1 lo
  g1[7] = 0;                                          // stride1 hi
  const int32x4 z4 = {0, 0, 0, 0};
#if __clang_major__ >= 23
  const int32x8 z8 = {0, 0, 0, 0, 0, 0, 0, 0};
  __builtin_amdgcn_tensor_load_to_lds(g0, g1, z4, z4, z8, 0);
#else
  __builtin_amdgcn_tensor_load_to_lds(g0, g1, z4, z4, 0);
#endif
}
#endif  // HAVE_TDM

// ---------------------------------------------------------------- converts
__global__ void cvt_f32_to_bf16(const float* __restrict__ in,
                                u16* __restrict__ out, int n) {
  int i = blockIdx.x * blockDim.x + threadIdx.x;
  if (i < n) out[i] = f2bf(in[i]);
}

// ---------------------------------------------------------------- QKV GEMM
// qkv[m, f] = sum_k x[m,k] * w_qkv[f,k] + b_qkv[f];  M=4096, N=3072, K=1024.
// Routes into q (b,h,s,d), k' = k/8 + rel (b,h,s,d), v^T (b,h,d,s), bf16.
__global__ __launch_bounds__(128) void qkv_gemm_kernel(
    const u16* __restrict__ xb, const u16* __restrict__ wq,
    const float* __restrict__ bqkv, const float* __restrict__ rel,
    u16* __restrict__ q, u16* __restrict__ kp, u16* __restrict__ vt) {
  const int lane = threadIdx.x & 31;
  const int wave = threadIdx.x >> 5;
  const int m0 = blockIdx.y * 64 + wave * 16;
  const int f0 = blockIdx.x * 64;

  v8f acc[4];
  const v8f z = {0.f, 0.f, 0.f, 0.f, 0.f, 0.f, 0.f, 0.f};
#pragma unroll
  for (int t = 0; t < 4; ++t) acc[t] = z;

  for (int k0 = 0; k0 < EE; k0 += 32) {
    v16bf a = frag_ld(xb, m0, EE, k0, lane);
#pragma unroll
    for (int t = 0; t < 4; ++t) {
      v16bf b = frag_ld(wq, f0 + 16 * t, EE, k0, lane);
      acc[t] = wmma_bf16(a, b, acc[t]);
    }
  }

  const int half = lane >> 4, col = lane & 15;
#pragma unroll
  for (int t = 0; t < 4; ++t) {
    const int f = f0 + 16 * t + col;
    const float bias = bqkv[f];
    const int h = f / (3 * HD);
    const int c = f % (3 * HD);
#pragma unroll
    for (int r = 0; r < 8; ++r) {
      const int m = m0 + r + 8 * half;
      const int b = m >> 11;
      const int s = m & (SS - 1);
      const float val = acc[t][r] + bias;
      if (c < HD) {
        q[((size_t)(b * HH + h) * SS + s) * HD + c] = f2bf(val);
      } else if (c < 2 * HD) {
        const int d = c - HD;
        const float rv = rel[((size_t)h * SS + s) * HD + d];
        kp[((size_t)(b * HH + h) * SS + s) * HD + d] = f2bf(val * 0.125f + rv);
      } else {
        const int d = c - 2 * HD;
        vt[((size_t)(b * HH + h) * HD + d) * SS + s] = f2bf(val);
      }
    }
  }
}

// ---------------------------------------------------------------- attention
// Per (b,h) / 64-row i-block: k' and v^T 64x64 tiles are staged in LDS by the
// Tensor Data Mover (double-buffered, wave 0 issues, s_wait_tensorcnt +
// barrier), logits = q@k'^T via bf16 WMMA from LDS fragments, online softmax,
// av = P@v, then sweep 2 recomputes logits and writes normalized weights once.
__global__ __launch_bounds__(128) void attention_kernel(
    const u16* __restrict__ q, const u16* __restrict__ kp,
    const u16* __restrict__ vt, u16* __restrict__ av,
    float* __restrict__ weights) {
  __shared__ u16 sTiles[2][2 * 64 * 64];  // [buf][ k-tile(64x64) | v-tile(64x64) ]
  __shared__ u16 sP[4 * 16 * 64];         // per-wave P staging (C->A relayout)

  const int lane = threadIdx.x & 31;
  const int wave = threadIdx.x >> 5;
  const int half = lane >> 4, col = lane & 15;
  const int bh = blockIdx.y;
  const int iw = blockIdx.x * 64 + wave * 16;
  const int NT = SS / 64;

  const u16* qh = q + (size_t)bh * SS * HD;
  const u16* kh = kp + (size_t)bh * SS * HD;
  const u16* vh = vt + (size_t)bh * HD * SS;
  u16* myP = &sP[wave * 16 * 64];

#if HAVE_TDM
  // Generic shared pointer low 32 bits == LDS byte offset (LDS aperture).
  const unsigned ldsK0 = (unsigned)(uintptr_t)&sTiles[0][0];
  const unsigned ldsK1 = (unsigned)(uintptr_t)&sTiles[1][0];
#endif

#if HAVE_TDM
#define ISSUE_TILES(it, withV)                                              \
  do {                                                                      \
    const unsigned lk = ((it) & 1) ? ldsK1 : ldsK0;                         \
    tdm_load_2d(kh + (size_t)(it) * 64 * HD, lk, 64, SS, 64, 64, HD);       \
    if (withV)                                                              \
      tdm_load_2d(vh + (size_t)(it) * 64, lk + 64 * 64 * 2, SS, HD, 64, 64, \
                  SS);                                                      \
  } while (0)
#else
  // Fallback: cooperative copy (2 half-rows per thread, b128 chunks).
#define ISSUE_TILES(it, withV)                                          \
  do {                                                                  \
    u16* kb_ = &sTiles[(it) & 1][0];                                    \
    u16* vb_ = &sTiles[(it) & 1][64 * 64];                              \
    const int row_ = threadIdx.x >> 1, part_ = (threadIdx.x & 1) * 32;  \
    const u16* ks_ = kh + (size_t)((it) * 64 + row_) * HD + part_;      \
    u16* kd_ = kb_ + row_ * 64 + part_;                                 \
    ((u32x4*)kd_)[0] = ((const u32x4*)ks_)[0];                          \
    ((u32x4*)kd_)[1] = ((const u32x4*)ks_)[1];                          \
    ((u32x4*)kd_)[2] = ((const u32x4*)ks_)[2];                          \
    ((u32x4*)kd_)[3] = ((const u32x4*)ks_)[3];                          \
    if (withV) {                                                        \
      const u16* vs_ = vh + (size_t)row_ * SS + (it) * 64 + part_;      \
      u16* vd_ = vb_ + row_ * 64 + part_;                               \
      ((u32x4*)vd_)[0] = ((const u32x4*)vs_)[0];                        \
      ((u32x4*)vd_)[1] = ((const u32x4*)vs_)[1];                        \
      ((u32x4*)vd_)[2] = ((const u32x4*)vs_)[2];                        \
      ((u32x4*)vd_)[3] = ((const u32x4*)vs_)[3];                        \
    }                                                                   \
  } while (0)
#endif

  // q fragments stay resident (16 rows x 64 K)
  const v16bf qf0 = frag_ld(qh, iw, HD, 0, lane);
  const v16bf qf1 = frag_ld(qh, iw, HD, 32, lane);

  float m[8], l[8];
  v8f accv[4];
  const v8f z = {0.f, 0.f, 0.f, 0.f, 0.f, 0.f, 0.f, 0.f};
#pragma unroll
  for (int r = 0; r < 8; ++r) { m[r] = -1e30f; l[r] = 0.f; }
#pragma unroll
  for (int t = 0; t < 4; ++t) accv[t] = z;

  // -------- sweep 1: online softmax + av accumulation --------
#if HAVE_TDM
  if (wave == 0) ISSUE_TILES(0, true);
#endif
  for (int it = 0; it < NT; ++it) {
    const int j0 = it * 64;
    const u16* kb = &sTiles[it & 1][0];
    const u16* vb = &sTiles[it & 1][64 * 64];
#if HAVE_TDM
    if (wave == 0) __builtin_amdgcn_s_wait_tensorcnt(0);
    __syncthreads();  // tiles ready; prev buffer fully consumed
    if (wave == 0 && it + 1 < NT) ISSUE_TILES(it + 1, true);
#else
    __syncthreads();
    ISSUE_TILES(it, true);
    __syncthreads();
#endif

    v8f f[4];
#pragma unroll
    for (int a = 0; a < 4; ++a) f[a] = z;
#pragma unroll
    for (int a = 0; a < 4; ++a) {
      v16bf b0 = frag_ld(kb, 16 * a, 64, 0, lane);
      f[a] = wmma_bf16(qf0, b0, f[a]);
      v16bf b1 = frag_ld(kb, 16 * a, 64, 32, lane);
      f[a] = wmma_bf16(qf1, b1, f[a]);
    }
    // row max (C layout: VGPR r -> row r + 8*half, 16 cols across lanes)
    float rm[8];
#pragma unroll
    for (int r = 0; r < 8; ++r)
      rm[r] = fmaxf(fmaxf(f[0][r], f[1][r]), fmaxf(f[2][r], f[3][r]));
#pragma unroll
    for (int d = 1; d < 16; d <<= 1)
#pragma unroll
      for (int r = 0; r < 8; ++r)
        rm[r] = fmaxf(rm[r], __shfl_xor(rm[r], d, 32));

    float mn[8], sc[8], ps[8];
#pragma unroll
    for (int r = 0; r < 8; ++r) {
      mn[r] = fmaxf(m[r], rm[r]);
      sc[r] = __expf(m[r] - mn[r]);
      ps[r] = 0.f;
    }
#pragma unroll
    for (int a = 0; a < 4; ++a)
#pragma unroll
      for (int r = 0; r < 8; ++r) {
        float p = __expf(f[a][r] - mn[r]);
        ps[r] += p;
        myP[(r + 8 * half) * 64 + a * 16 + col] = f2bf(p);
      }
#pragma unroll
    for (int d = 1; d < 16; d <<= 1)
#pragma unroll
      for (int r = 0; r < 8; ++r)
        ps[r] += __shfl_xor(ps[r], d, 32);
#pragma unroll
    for (int r = 0; r < 8; ++r) {
      l[r] = l[r] * sc[r] + ps[r];
      m[r] = mn[r];
    }
#pragma unroll
    for (int t = 0; t < 4; ++t)
#pragma unroll
      for (int r = 0; r < 8; ++r) accv[t][r] *= sc[r];

    // av += P @ v   (A = P from LDS, B = staged v^T tile, (N=d, K=j_local))
    v16bf p0 = frag_ld(myP, 0, 64, 0, lane);
    v16bf p1 = frag_ld(myP, 0, 64, 32, lane);
#pragma unroll
    for (int t = 0; t < 4; ++t) {
      v16bf bv0 = frag_ld(vb, 16 * t, 64, 0, lane);
      accv[t] = wmma_bf16(p0, bv0, accv[t]);
      v16bf bv1 = frag_ld(vb, 16 * t, 64, 32, lane);
      accv[t] = wmma_bf16(p1, bv1, accv[t]);
    }
    __syncthreads();  // done reading this buffer
    (void)j0;
  }

  float inv[8];
#pragma unroll
  for (int r = 0; r < 8; ++r) inv[r] = 1.0f / l[r];

#if HAVE_TDM
  if (wave == 0) ISSUE_TILES(0, false);  // prologue for sweep 2
#endif

  // write av (bf16, (b, s, h*HD+d) layout for the output projection GEMM)
  const int bq = bh >> 4, hh = bh & (HH - 1);
#pragma unroll
  for (int t = 0; t < 4; ++t)
#pragma unroll
    for (int r = 0; r < 8; ++r) {
      const int i = iw + r + 8 * half;
      av[((size_t)(bq * SS + i)) * EE + hh * HD + t * 16 + col] =
          f2bf(accv[t][r] * inv[r]);
    }

  // -------- sweep 2: recompute logits, write normalized weights --------
  for (int it = 0; it < NT; ++it) {
    const int j0 = it * 64;
    const u16* kb = &sTiles[it & 1][0];
#if HAVE_TDM
    if (wave == 0) __builtin_amdgcn_s_wait_tensorcnt(0);
    __syncthreads();
    if (wave == 0 && it + 1 < NT) ISSUE_TILES(it + 1, false);
#else
    __syncthreads();
    ISSUE_TILES(it, false);
    __syncthreads();
#endif

    v8f f[4];
#pragma unroll
    for (int a = 0; a < 4; ++a) f[a] = z;
#pragma unroll
    for (int a = 0; a < 4; ++a) {
      v16bf b0 = frag_ld(kb, 16 * a, 64, 0, lane);
      f[a] = wmma_bf16(qf0, b0, f[a]);
      v16bf b1 = frag_ld(kb, 16 * a, 64, 32, lane);
      f[a] = wmma_bf16(qf1, b1, f[a]);
    }
#pragma unroll
    for (int a = 0; a < 4; ++a)
#pragma unroll
      for (int r = 0; r < 8; ++r) {
        const int i = iw + r + 8 * half;
        const float wv = __expf(f[a][r] - m[r]) * inv[r];
        weights[((size_t)bh * SS + i) * SS + (j0 + a * 16 + col)] = wv;
      }
    __syncthreads();
  }
#undef ISSUE_TILES
}

// ---------------------------------------------------------------- out proj
// out[m, f] = sum_k av[m,k] * w_out[f,k] + b_out[f];  M=4096, N=K=1024
__global__ __launch_bounds__(128) void out_gemm_kernel(
    const u16* __restrict__ avb, const u16* __restrict__ wo,
    const float* __restrict__ bo, float* __restrict__ out) {
  const int lane = threadIdx.x & 31;
  const int wave = threadIdx.x >> 5;
  const int m0 = blockIdx.y * 64 + wave * 16;
  const int f0 = blockIdx.x * 64;

  v8f acc[4];
  const v8f z = {0.f, 0.f, 0.f, 0.f, 0.f, 0.f, 0.f, 0.f};
#pragma unroll
  for (int t = 0; t < 4; ++t) acc[t] = z;

  for (int k0 = 0; k0 < EE; k0 += 32) {
    v16bf a = frag_ld(avb, m0, EE, k0, lane);
#pragma unroll
    for (int t = 0; t < 4; ++t) {
      v16bf b = frag_ld(wo, f0 + 16 * t, EE, k0, lane);
      acc[t] = wmma_bf16(a, b, acc[t]);
    }
  }

  const int half = lane >> 4, col = lane & 15;
#pragma unroll
  for (int t = 0; t < 4; ++t) {
    const int f = f0 + 16 * t + col;
    const float bias = bo[f];
#pragma unroll
    for (int r = 0; r < 8; ++r) {
      const int mm = m0 + r + 8 * half;
      out[(size_t)mm * EE + f] = acc[t][r] + bias;
    }
  }
}

// ---------------------------------------------------------------- launcher
extern "C" void kernel_launch(void* const* d_in, const int* in_sizes, int n_in,
                              void* d_out, int out_size, void* d_ws,
                              size_t ws_size, hipStream_t stream) {
  const float* x     = (const float*)d_in[0];   // (B,S,E)
  const float* w_qkv = (const float*)d_in[1];   // (3E,E)
  const float* b_qkv = (const float*)d_in[2];   // (3E)
  const float* w_out = (const float*)d_in[3];   // (E,E)
  const float* b_out = (const float*)d_in[4];   // (E)
  const float* rel   = (const float*)d_in[5];   // (H,S,HD)

  float* out     = (float*)d_out;                       // (B,S,E)
  float* weights = out + (size_t)BB * SS * EE;          // (B,H,S,S)

  const size_t Q = (size_t)BB * HH * SS * HD;  // 4,194,304 elems (== B*S*E)
  u16* ws    = (u16*)d_ws;
  u16* q     = ws;
  u16* kp    = q + Q;
  u16* vt    = kp + Q;
  u16* avb   = vt + Q;
  u16* xb    = avb + Q;
  u16* wqkvb = xb + Q;
  u16* woutb = wqkvb + (size_t)3 * EE * EE;

  // 1) fp32 -> bf16 converts
  {
    int n = BB * SS * EE;
    cvt_f32_to_bf16<<<(n + 255) / 256, 256, 0, stream>>>(x, xb, n);
    n = 3 * EE * EE;
    cvt_f32_to_bf16<<<(n + 255) / 256, 256, 0, stream>>>(w_qkv, wqkvb, n);
    n = EE * EE;
    cvt_f32_to_bf16<<<(n + 255) / 256, 256, 0, stream>>>(w_out, woutb, n);
  }

  // 2) QKV projection -> q, k' = k/8 + rel, v^T (bf16)
  {
    dim3 grid(3 * EE / 64, (BB * SS) / 64);   // (48, 64)
    qkv_gemm_kernel<<<grid, 128, 0, stream>>>(xb, wqkvb, b_qkv, rel, q, kp, vt);
  }

  // 3) fused attention: TDM-staged tiles, av (bf16) + weights (fp32)
  {
    dim3 grid(SS / 64, BB * HH);              // (32, 32)
    attention_kernel<<<grid, 128, 0, stream>>>(q, kp, vt, avb, weights);
  }

  // 4) output projection
  {
    dim3 grid(EE / 64, (BB * SS) / 64);       // (16, 64)
    out_gemm_kernel<<<grid, 128, 0, stream>>>(avb, woutb, b_out, out);
  }
}
